// MFGCGRU_SM_23570780521077
// MI455X (gfx1250) — matrix operand order repaired and checked by reading, not compiled, not run on hardware
//
#include <hip/hip_runtime.h>
#include <hip/hip_bf16.h>
#include <stdint.h>

// ---------------- problem constants ----------------
#define BB   32
#define NN   2048
#define DIN  2
#define UU   64
#define FD   32
#define SD   64
#define MM   4
#define ROWS 66            // DIN + U
#define COLS (BB * ROWS)   // 2112 columns of the propagation GEMM

typedef _Float16 half8  __attribute__((ext_vector_type(8)));
typedef _Float16 half16 __attribute__((ext_vector_type(16)));
typedef float    f32x8  __attribute__((ext_vector_type(8)));

// ---------------- WMMA fragment helpers (CDNA5 16x16x32 f16, wave32) -------
// A (16x32 f16): lane<16 -> row=lane, k=[0..15];
//                lane>=16 -> row=lane-16, k=[16..31]
static __device__ inline half16 loadA16x32(const _Float16* __restrict__ base,
                                           int ld, int row0, int k0, int lane) {
  int row = row0 + (lane & 15);
  int kh  = (lane >> 4) * 16;
  const _Float16* p = base + (size_t)row * ld + k0 + kh;
  half8 lo = *(const half8*)p;
  half8 hi = *(const half8*)(p + 8);
  return __builtin_shufflevector(lo, hi, 0,1,2,3,4,5,6,7,8,9,10,11,12,13,14,15);
}

// A, scalar-load variant (unaligned row bases, cold path)
static __device__ inline half16 loadA16x32_s(const _Float16* __restrict__ base,
                                             int ld, int row0, int k0, int lane) {
  int row = row0 + (lane & 15);
  int kh  = (lane >> 4) * 16;
  const _Float16* p = base + (size_t)row * ld + k0 + kh;
  half16 a;
#pragma unroll
  for (int e = 0; e < 8; ++e) { a[e] = p[e]; a[8 + e] = p[8 + e]; }
  return a;
}

// B (32x16 f16) from storage where element [k][col] lives at src[col*ld + k]:
// lane<16 -> col=lane, k=[0..15]; lane>=16 -> col=lane-16, k=[16..31]
static __device__ inline half16 loadB32x16(const _Float16* __restrict__ base,
                                           int ld, int col0, int k0, int lane) {
  int col = col0 + (lane & 15);
  int kh  = (lane >> 4) * 16;
  return *(const half16*)(base + (size_t)col * ld + k0 + kh);
}

// B from row-major [k][ncols] weights (strided, cold path)
static __device__ inline half16 loadB32x16_rm(const _Float16* __restrict__ base,
                                              int ncols, int col0, int k0, int lane) {
  int col = col0 + (lane & 15);
  int kh  = (lane >> 4) * 16;
  half16 b;
#pragma unroll
  for (int e = 0; e < 16; ++e) b[e] = base[(size_t)(k0 + kh + e) * ncols + col];
  return b;
}

static __device__ inline f32x8 wmma(half16 a, half16 b, f32x8 c) {
  return __builtin_amdgcn_wmma_f32_16x16x32_f16(false, a, false, b, (short)0, c,
                                                false, false);
}

// CDNA5 async global->LDS copy (ASYNCcnt-tracked, bypasses VGPRs)
static __device__ inline void async_ldscpy_b128(uint32_t lds_off,
                                                const void* gaddr) {
  asm volatile("global_load_async_to_lds_b128 %0, %1, off"
               :: "v"(lds_off), "v"(gaddr) : "memory");
}
static __device__ inline void wait_async0() {
  asm volatile("s_wait_asynccnt 0x0" ::: "memory");
}

// ---------------- kernel 1: Q, K (relu, fp16) and sentinel s ----------------
__global__ void k_prep(const float* __restrict__ feat, const float* __restrict__ SE,
                       const float* __restrict__ Wq, const float* __restrict__ Wk,
                       const float* __restrict__ Ws1, const float* __restrict__ bs1,
                       const float* __restrict__ Ws2, const float* __restrict__ bs2,
                       _Float16* __restrict__ Qh, _Float16* __restrict__ Kh,
                       float* __restrict__ sbuf) {
  int n = blockIdx.x * blockDim.x + threadIdx.x;   // 2048 threads exactly
  const float* f = feat + (size_t)n * FD;
  const float* se = SE + (size_t)n * SD;
  for (int u = 0; u < UU; ++u) {
    float q = 0.f, k = 0.f;
    for (int i = 0; i < FD; ++i) {
      float fv = f[i];
      q += fv * Wq[i * UU + u];
      k += fv * Wk[i * UU + u];
    }
    Qh[(size_t)n * UU + u] = (_Float16)fmaxf(q, 0.f);
    Kh[(size_t)n * UU + u] = (_Float16)fmaxf(k, 0.f);
  }
  float s = 0.f;
  for (int u = 0; u < UU; ++u) {
    float h = bs1[u];
    for (int i = 0; i < FD; ++i) h += f[i]  * Ws1[i * UU + u];
    for (int j = 0; j < SD; ++j) h += se[j] * Ws1[(FD + j) * UU + u];
    s += fmaxf(h, 0.f) * Ws2[u];
  }
  sbuf[n] = fmaxf(s + bs2[0], 0.f);
}

// ---------------- kernel 2: adj1/adj2 -> fp16 -------------------------------
__global__ void k_cvt_adj(const float* __restrict__ a1, const float* __restrict__ a2,
                          _Float16* __restrict__ S) {
  size_t i = (size_t)blockIdx.x * blockDim.x + threadIdx.x;  // 2*N*N exactly
  const size_t NSQ = (size_t)NN * NN;
  float v = (i < NSQ) ? a1[i] : a2[i - NSQ];
  S[i] = (_Float16)v;
}

// ---------------- kernel 3: gate kernels -> fp16 ----------------------------
__global__ void k_cvt_w(const float* __restrict__ rk, const float* __restrict__ uk,
                        const float* __restrict__ ck, _Float16* __restrict__ kw) {
  int i = blockIdx.x * blockDim.x + threadIdx.x;   // 3*4*66*64 exactly
  const int per = MM * ROWS * UU;                  // 16896
  int g = i / per, rem = i - g * per;
  const float* src = (g == 0) ? rk : (g == 1) ? uk : ck;
  kw[i] = (_Float16)src[rem];
}

__global__ void k_bmean(const float* __restrict__ rb, const float* __restrict__ ub,
                        const float* __restrict__ cb, float* __restrict__ bm) {
  int t = threadIdx.x;
  if (t < 3 * UU) {
    int g = t / UU, u = t - g * UU;
    const float* b = (g == 0) ? rb : (g == 1) ? ub : cb;
    bm[t] = 0.25f * (b[u] + b[UU + u] + b[2 * UU + u] + b[3 * UU + u]);
  }
}

// ---------------- kernel 4: attention support (two-pass WMMA) ---------------
// one wave per 16-row strip; pass1 rowsums, pass2 normalize + fp16 store
__global__ void k_att(const _Float16* __restrict__ Qh, const _Float16* __restrict__ Kh,
                      const float* __restrict__ sbuf, _Float16* __restrict__ Satt) {
  __shared__ float red[8][32][8];
  int tid = threadIdx.x, wave = tid >> 5, lane = tid & 31;
  int strip = blockIdx.x * 8 + wave;
  int row0 = strip * 16;
  int hi = lane >> 4, nlo = lane & 15;

  half16 a0 = loadA16x32(Qh, UU, row0, 0, lane);
  half16 a1 = loadA16x32(Qh, UU, row0, 32, lane);

  f32x8 rs = {};
  for (int ct = 0; ct < NN / 16; ++ct) {
    f32x8 c = {};
    half16 b0 = loadB32x16(Kh, UU, ct * 16, 0, lane);
    half16 b1 = loadB32x16(Kh, UU, ct * 16, 32, lane);
    c = wmma(a0, b0, c);
    c = wmma(a1, b1, c);
#pragma unroll
    for (int r = 0; r < 8; ++r) rs[r] += __expf(c[r] * 0.125f);
  }
#pragma unroll
  for (int r = 0; r < 8; ++r) red[wave][lane][r] = rs[r];
  __syncthreads();

  float inv[8];
#pragma unroll
  for (int r = 0; r < 8; ++r) {
    float d = sbuf[row0 + r + 8 * hi];
    for (int l = 0; l < 16; ++l) d += red[wave][hi * 16 + l][r];
    inv[r] = 1.0f / d;
  }

  for (int ct = 0; ct < NN / 16; ++ct) {
    f32x8 c = {};
    half16 b0 = loadB32x16(Kh, UU, ct * 16, 0, lane);
    half16 b1 = loadB32x16(Kh, UU, ct * 16, 32, lane);
    c = wmma(a0, b0, c);
    c = wmma(a1, b1, c);
#pragma unroll
    for (int r = 0; r < 8; ++r) {
      float e = __expf(c[r] * 0.125f) * inv[r];
      Satt[(size_t)(row0 + r + 8 * hi) * NN + ct * 16 + nlo] = (_Float16)e;
    }
  }
}

// ---------------- kernel 5: build X in both layouts -------------------------
// mode 0: state = h ; mode 1: state = r*h
__global__ void k_concat(const float* __restrict__ inputs, const float* __restrict__ h,
                         const float* __restrict__ rbuf, int mode,
                         _Float16* __restrict__ Xn, _Float16* __restrict__ Xt) {
  size_t i = (size_t)blockIdx.x * blockDim.x + threadIdx.x;  // COLS*NN exactly
  int col = (int)(i >> 11);
  int n   = (int)(i & (NN - 1));
  int b = col / ROWS;
  int f = col - b * ROWS;
  float v;
  if (f < DIN) {
    v = inputs[((size_t)b * NN + n) * DIN + f];
  } else {
    size_t hidx = ((size_t)b * NN + n) * UU + (f - DIN);
    v = h[hidx];
    if (mode == 1) v *= rbuf[hidx];
  }
  _Float16 hv = (_Float16)v;
  Xt[(size_t)col * NN + n]   = hv;   // [col][node] : B-operand layout (coalesced)
  Xn[(size_t)n * COLS + col] = hv;   // [node][col] : A-operand layout for proj
}

// ---------------- kernel 6: the big support GEMM ----------------------------
// Y[m][n][col] = sum_k S_m[n][k] * X[k][col];  block = 128 rows x 64 cols.
// B strip (64 cols x 32 k) staged via double-buffered async global->LDS copy:
// iteration kt waits its own ASYNCcnt, barriers (closes the WAR on the buffer
// being refilled), prefetches strip kt+1, then runs WMMAs out of strip kt
// while the async engine fills the other buffer.
__global__ void k_gemmS(const _Float16* __restrict__ S, const _Float16* __restrict__ Xt,
                        _Float16* __restrict__ Y) {
  __shared__ _Float16 lbs[2][64 * 32];            // 2 x 4 KB
  int tid = threadIdx.x, wave = tid >> 5, lane = tid & 31;
  int m  = blockIdx.z;
  int nb = blockIdx.y;
  int cs = blockIdx.x;
  const _Float16* Sm = S + (size_t)m * NN * NN;
  int row0 = nb * 128 + wave * 16;
  int col0 = cs * 64;
  int hi = lane >> 4, nlo = lane & 15;

  // per-thread async-copy slot: 16 bytes of one LDS column
  int ccol = tid >> 2;
  int seg  = (tid & 3) * 8;
  const _Float16* gsrc = Xt + (size_t)(col0 + ccol) * NN + seg;
  uint32_t lds0 = (uint32_t)(uintptr_t)&lbs[0][ccol * 32 + seg];
  uint32_t lds1 = (uint32_t)(uintptr_t)&lbs[1][ccol * 32 + seg];

  async_ldscpy_b128(lds0, gsrc);                  // prologue: stage strip 0

  f32x8 c0 = {}, c1 = {}, c2 = {}, c3 = {};
  for (int kt = 0; kt < NN / 32; ++kt) {
    wait_async0();                                // my strip-kt bytes landed
    __syncthreads();                              // everyone's landed; prev reads done
    if (kt + 1 < NN / 32) {                       // uniform branch
      async_ldscpy_b128((kt & 1) ? lds0 : lds1, gsrc + (size_t)(kt + 1) * 32);
    }
    const _Float16* cur = lbs[kt & 1];
    half16 a = loadA16x32(Sm, NN, row0, kt * 32, lane);
    c0 = wmma(a, loadB32x16(cur, 32,  0, 0, lane), c0);
    c1 = wmma(a, loadB32x16(cur, 32, 16, 0, lane), c1);
    c2 = wmma(a, loadB32x16(cur, 32, 32, 0, lane), c2);
    c3 = wmma(a, loadB32x16(cur, 32, 48, 0, lane), c3);
  }
  _Float16* Ym = Y + (size_t)m * NN * COLS;
#pragma unroll
  for (int r = 0; r < 8; ++r) {
    size_t rowoff = (size_t)(row0 + r + 8 * hi) * COLS + col0 + nlo;
    Ym[rowoff +  0] = (_Float16)c0[r];
    Ym[rowoff + 16] = (_Float16)c1[r];
    Ym[rowoff + 32] = (_Float16)c2[r];
    Ym[rowoff + 48] = (_Float16)c3[r];
  }
}

// ---------------- kernel 7: projection + gate nonlinearity ------------------
// mode 0: r = sigmoid(.) -> outbuf ; mode 1: u = sigmoid(.) -> outbuf
// mode 2: c = tanh(.), h_new = u*h + (1-u)*c -> outbuf (d_out)
__global__ void k_proj(const _Float16* __restrict__ Xn, const _Float16* __restrict__ Y,
                       const _Float16* __restrict__ kw,   // [4][66][64] fp16
                       const float* __restrict__ bm,      // [64]
                       const float* __restrict__ ubuf, const float* __restrict__ h,
                       float* __restrict__ outbuf, int mode) {
  int tid = threadIdx.x, lane = tid & 31;
  int task = blockIdx.x * 8 + (tid >> 5);        // 4096 tasks exactly
  int b  = task >> 7;
  int nt = task & 127;
  int row0 = nt * 16;
  int hi = lane >> 4, nlo = lane & 15;

  f32x8 c[4] = {};
  for (int m = 0; m < MM; ++m) {
    const _Float16* A = (m == 0) ? Xn : (Y + (size_t)(m - 1) * NN * COLS);
    half16 a0 = loadA16x32_s(A, COLS, row0, b * ROWS,      lane);
    half16 a1 = loadA16x32_s(A, COLS, row0, b * ROWS + 32, lane);
    const _Float16* km = kw + (size_t)m * ROWS * UU;
    float k64[4], k65[4];
#pragma unroll
    for (int j = 0; j < 4; ++j) {
      int u = j * 16 + nlo;
      k64[j] = (float)km[64 * UU + u];
      k65[j] = (float)km[65 * UU + u];
    }
#pragma unroll
    for (int j = 0; j < 4; ++j) {
      c[j] = wmma(a0, loadB32x16_rm(km, UU, j * 16, 0,  lane), c[j]);
      c[j] = wmma(a1, loadB32x16_rm(km, UU, j * 16, 32, lane), c[j]);
    }
#pragma unroll
    for (int r = 0; r < 8; ++r) {             // exact tail for f = 64, 65
      int row = row0 + r + 8 * hi;
      float x64 = (float)A[(size_t)row * COLS + b * ROWS + 64];
      float x65 = (float)A[(size_t)row * COLS + b * ROWS + 65];
#pragma unroll
      for (int j = 0; j < 4; ++j) c[j][r] += x64 * k64[j] + x65 * k65[j];
    }
  }
#pragma unroll
  for (int j = 0; j < 4; ++j) {
    int u = j * 16 + nlo;
    float bias = bm[u];
#pragma unroll
    for (int r = 0; r < 8; ++r) {
      int row = row0 + r + 8 * hi;
      float v = c[j][r] * 0.25f + bias;
      size_t idx = ((size_t)b * NN + row) * UU + u;
      if (mode < 2) {
        outbuf[idx] = 1.0f / (1.0f + __expf(-v));
      } else {
        float cc = tanhf(v);
        float uu = ubuf[idx];
        outbuf[idx] = uu * h[idx] + (1.0f - uu) * cc;
      }
    }
  }
}

// ---------------- host-side orchestration -----------------------------------
extern "C" void kernel_launch(void* const* d_in, const int* in_sizes, int n_in,
                              void* d_out, int out_size, void* d_ws, size_t ws_size,
                              hipStream_t stream) {
  const float* inputs = (const float*)d_in[0];
  const float* h_prev = (const float*)d_in[1];
  const float* adj1   = (const float*)d_in[2];
  const float* adj2   = (const float*)d_in[3];
  const float* feat   = (const float*)d_in[4];
  const float* SE     = (const float*)d_in[5];
  const float* Wq     = (const float*)d_in[6];
  const float* Wk     = (const float*)d_in[7];
  const float* Ws1    = (const float*)d_in[8];
  const float* bs1    = (const float*)d_in[9];
  const float* Ws2    = (const float*)d_in[10];
  const float* bs2    = (const float*)d_in[11];
  const float* r_kernel = (const float*)d_in[12];
  const float* r_bias   = (const float*)d_in[13];
  const float* u_kernel = (const float*)d_in[14];
  const float* u_bias   = (const float*)d_in[15];
  const float* c_kernel = (const float*)d_in[16];
  const float* c_bias   = (const float*)d_in[17];

  char* ws = (char*)d_ws;
  size_t off = 0;
  auto alloc = [&](size_t bytes) -> char* {
    char* p = ws + off;
    off += (bytes + 255) & ~(size_t)255;
    return p;
  };
  _Float16* Qh   = (_Float16*)alloc((size_t)NN * UU * 2);
  _Float16* Kh   = (_Float16*)alloc((size_t)NN * UU * 2);
  float*    sbuf = (float*)   alloc((size_t)NN * 4);
  _Float16* Sf   = (_Float16*)alloc((size_t)3 * NN * NN * 2);   // [adj1,adj2,Satt]
  _Float16* Xn   = (_Float16*)alloc((size_t)NN * COLS * 2);
  _Float16* Xt   = (_Float16*)alloc((size_t)COLS * NN * 2);
  _Float16* Yb   = (_Float16*)alloc((size_t)3 * NN * COLS * 2);
  _Float16* kwf  = (_Float16*)alloc((size_t)3 * MM * ROWS * UU * 2);
  float*    bm   = (float*)   alloc((size_t)3 * UU * 4);
  float*    rbuf = (float*)   alloc((size_t)BB * NN * UU * 4);
  float*    ubuf = (float*)   alloc((size_t)BB * NN * UU * 4);
  float*    out  = (float*)d_out;

  const int per = MM * ROWS * UU;  // per-gate weight count

  // small prep / conversions
  k_prep<<<NN / 256, 256, 0, stream>>>(feat, SE, Wq, Wk, Ws1, bs1, Ws2, bs2,
                                       Qh, Kh, sbuf);
  k_cvt_adj<<<(2 * NN * NN) / 256, 256, 0, stream>>>(adj1, adj2, Sf);
  k_cvt_w<<<(3 * per) / 256, 256, 0, stream>>>(r_kernel, u_kernel, c_kernel, kwf);
  k_bmean<<<1, 256, 0, stream>>>(r_bias, u_bias, c_bias, bm);

  // attention support -> Sf[2]
  k_att<<<NN / 128, 256, 0, stream>>>(Qh, Kh, sbuf, Sf + (size_t)2 * NN * NN);

  // round 1: X = [inputs, h]
  k_concat<<<(COLS * NN) / 256, 256, 0, stream>>>(inputs, h_prev, rbuf, 0, Xn, Xt);
  k_gemmS<<<dim3(COLS / 64, NN / 128, 3), 256, 0, stream>>>(Sf, Xt, Yb);
  k_proj<<<(BB * NN / 16) / 8, 256, 0, stream>>>(Xn, Yb, kwf + 0 * per, bm + 0 * UU,
                                                 ubuf, h_prev, rbuf, 0);
  k_proj<<<(BB * NN / 16) / 8, 256, 0, stream>>>(Xn, Yb, kwf + 1 * per, bm + 1 * UU,
                                                 ubuf, h_prev, ubuf, 1);

  // round 2: X = [inputs, r*h]
  k_concat<<<(COLS * NN) / 256, 256, 0, stream>>>(inputs, h_prev, rbuf, 1, Xn, Xt);
  k_gemmS<<<dim3(COLS / 64, NN / 128, 3), 256, 0, stream>>>(Sf, Xt, Yb);
  k_proj<<<(BB * NN / 16) / 8, 256, 0, stream>>>(Xn, Yb, kwf + 2 * per, bm + 2 * UU,
                                                 ubuf, h_prev, out, 2);
}
